// EncoderBlock_47528108097844
// MI455X (gfx1250) — compile-verified, matrix operand order checked by reading, not run on hardware
//
#include <hip/hip_runtime.h>
#include <hip/hip_bf16.h>

typedef __attribute__((ext_vector_type(16))) _Float16 v16h;
typedef __attribute__((ext_vector_type(8)))  _Float16 h8;
typedef __attribute__((ext_vector_type(8)))  float    v8f;

#define TOK   16384   // B*A*N tokens
#define HD    256
#define NSEQ  1024
#define DHEAD 32
#define FFND  512

static __device__ __forceinline__ v8f wmma16(v16h a, v16h b, v8f c) {
  // D(16x16,f32) = A(16x32,f16) x B(32x16,f16) + C
  return __builtin_amdgcn_wmma_f32_16x16x32_f16(false, a, false, b, (short)0, c,
                                                false, false);
}

// Load one 16x32 f16 operand fragment (A layout; Bt uses the same layout with
// row = output column). Source is row-major [rows, ld] f16, 16B-aligned rows.
// Lane L holds matrix row (row0 + L%16); its 16 halves cover
// K = k0 + i + (i&8) + ((L&16)>>1)  ->  two contiguous groups of 8 halves,
// fetched as two global_load_b128 / ds_load_b128.
static __device__ __forceinline__ v16h load_frag(const _Float16* base, int ld,
                                                 int row0, int k0) {
  int lane = threadIdx.x & 31;
  int r    = row0 + (lane & 15);
  int koff = (lane & 16) >> 1;           // 0 or 8
  const _Float16* p = base + (size_t)r * ld + k0 + koff;
  h8 g0 = *(const h8*)(p);
  h8 g1 = *(const h8*)(p + 16);
  v16h out;
#pragma unroll
  for (int i = 0; i < 8; ++i) { out[i] = g0[i]; out[i + 8] = g1[i]; }
  return out;
}

// ---------------- format conversion kernels ----------------

__global__ void __launch_bounds__(256)
k_f32_to_f16(const float* __restrict__ in, _Float16* __restrict__ out, int n) {
  int i = blockIdx.x * blockDim.x + threadIdx.x;
  if (i < n) out[i] = (_Float16)in[i];
}

// in: f32 [K,N] row-major  ->  out: f16 [N,K] (B-transposed operand)
__global__ void __launch_bounds__(256)
k_transpose_to_f16(const float* __restrict__ in, _Float16* __restrict__ out,
                   int K, int N) {
  int i = blockIdx.x * blockDim.x + threadIdx.x;
  if (i >= K * N) return;
  int k = i / N, n = i - k * N;
  out[n * K + k] = (_Float16)in[i];
}

// ---------------- fused QKV projection ----------------
// Q/K written in [bh][a][n][d] layout (the _transpose_qkv permutation),
// V written transposed per slice: Vt[bh][a][d][n].
__global__ void __launch_bounds__(256)
k_qkv(const _Float16* __restrict__ Xh,
      const _Float16* __restrict__ WqT, const _Float16* __restrict__ WkT,
      const _Float16* __restrict__ WvT,
      _Float16* __restrict__ Qb, _Float16* __restrict__ Kb,
      _Float16* __restrict__ Vtb) {
  int wave = (blockIdx.x * blockDim.x + threadIdx.x) >> 5;   // 16384 waves
  int mt = wave >> 4;          // 1024 row tiles of 16 tokens
  int nt = wave & 15;          // 16 col tiles of 16
  int lane = threadIdx.x & 31;
  v8f aq = {}, ak = {}, av = {};
#pragma unroll
  for (int k0 = 0; k0 < HD; k0 += 32) {
    v16h af = load_frag(Xh, HD, mt * 16, k0);
    v16h bq = load_frag(WqT, HD, nt * 16, k0); aq = wmma16(af, bq, aq);
    v16h bk = load_frag(WkT, HD, nt * 16, k0); ak = wmma16(af, bk, ak);
    v16h bv = load_frag(WvT, HD, nt * 16, k0); av = wmma16(af, bv, av);
  }
  int col = nt * 16 + (lane & 15);
  int hh = col >> 5, d = col & 31;
  int rbase = (lane & 16) >> 1;           // +8 for upper half-wave
#pragma unroll
  for (int r = 0; r < 8; ++r) {
    int t = mt * 16 + r + rbase;          // token id = ((b*4+a)*1024+n)
    int n = t & 1023, a = (t >> 10) & 3, b = t >> 12;
    int bh = b * 8 + hh;
    int sl = bh * 4 + a;                  // slice in [0,128)
    Qb[((size_t)(sl << 10) + n) * DHEAD + d] = (_Float16)aq[r];
    Kb[((size_t)(sl << 10) + n) * DHEAD + d] = (_Float16)ak[r];
    Vtb[((size_t)sl * DHEAD + d) * NSEQ + n] = (_Float16)av[r];
  }
}

// ---------------- masked attention, fused permuted output ----------------
// One wave = one 16-row Q tile of one slice. D=32 -> Q@K^T tile is one WMMA.
// Softmax without max-subtraction (scores are O(0.1); masked -> p=0).
// P tile bounced through LDS (C-layout -> A-layout) into att@V WMMAs.
__global__ void __launch_bounds__(256)
k_attn(const _Float16* __restrict__ Qb, const _Float16* __restrict__ Kb,
       const _Float16* __restrict__ Vtb, const int* __restrict__ adj,
       _Float16* __restrict__ outPerm) {
  __shared__ _Float16 pbuf[8][16 * 32];
  int wv = threadIdx.x >> 5;
  int lane = threadIdx.x & 31;
  int slice = blockIdx.x >> 3;            // 128 slices
  int qblk  = blockIdx.x & 7;
  int qn = qblk * 128 + wv * 16;          // Q tile origin within slice
  const _Float16* Qs = Qb  + (size_t)slice * NSEQ * DHEAD;
  const _Float16* Ks = Kb  + (size_t)slice * NSEQ * DHEAD;
  const _Float16* Vs = Vtb + (size_t)slice * DHEAD * NSEQ;
  _Float16* pb = pbuf[wv];
  int l15 = lane & 15;
  int rbase = (lane & 16) >> 1;

  v16h qf = load_frag(Qs, DHEAD, qn, 0);  // whole K=D=32 in one fragment
  v8f hp0 = {}, hp1 = {};                 // d 0..15 / 16..31 accumulators
  float lsum[8];
#pragma unroll
  for (int r = 0; r < 8; ++r) lsum[r] = 0.f;

  for (int kn = 0; kn < NSEQ; kn += 32) {
    v16h kf0 = load_frag(Ks, DHEAD, kn, 0);
    v16h kf1 = load_frag(Ks, DHEAD, kn + 16, 0);
    v8f z = {};
    v8f s0 = wmma16(qf, kf0, z);
    v8f s1 = wmma16(qf, kf1, z);
#pragma unroll
    for (int r = 0; r < 8; ++r) {
      int qg = qn + r + rbase;                       // query row in slice
      const int* arow = adj + (size_t)qg * NSEQ + kn;
      float p0 = (arow[l15]      > 0) ? __expf(s0[r] * 0.125f) : 0.0f;
      float p1 = (arow[16 + l15] > 0) ? __expf(s1[r] * 0.125f) : 0.0f;
      lsum[r] += p0 + p1;
      int m = r + rbase;                             // tile-local row
      pb[m * 32 + l15]      = (_Float16)p0;
      pb[m * 32 + 16 + l15] = (_Float16)p1;
    }
    asm volatile("s_wait_dscnt 0x0" ::: "memory");   // P tile visible in LDS
    v16h pf = load_frag(pb, 32, 0, 0);               // A-layout reload
    v16h v0 = load_frag(Vs, NSEQ, 0,  kn);           // Bt rows = d 0..15
    v16h v1 = load_frag(Vs, NSEQ, 16, kn);           // Bt rows = d 16..31
    hp0 = wmma16(pf, v0, hp0);
    hp1 = wmma16(pf, v1, hp1);
  }

#pragma unroll
  for (int r = 0; r < 8; ++r) {
    float s = lsum[r];
    s += __shfl_xor(s, 1); s += __shfl_xor(s, 2);
    s += __shfl_xor(s, 4); s += __shfl_xor(s, 8);    // row-sum over 16 lanes
    float rinv = 1.0f / (s + 1e-30f);
    // hp flat row R -> torch-faithful permuted (b,a,n,hh) position
    int R = slice * NSEQ + qn + r + rbase;           // [0, 131072)
    int k2 = R & 3, j2 = (R >> 2) & 31, i2 = R >> 7;
    int G = (k2 << 15) | (i2 << 5) | j2;
    int hh = G & 7, n2 = (G >> 3) & 1023, a2 = (G >> 13) & 3, b2 = G >> 15;
    size_t row = (size_t)(((b2 * 4 + a2) << 10) + n2);
    outPerm[row * HD + hh * 32 + l15]      = (_Float16)(hp0[r] * rinv);
    outPerm[row * HD + hh * 32 + 16 + l15] = (_Float16)(hp1[r] * rinv);
  }
}

// ---------------- generic WMMA GEMM with epilogue ----------------
// C[M,N] = A[M,K](f16) @ Bt[N,K](f16)^T ; +bias, ReLU, +resid(f32),
// optional f32 and/or f16 outputs. One wave -> 16x64 tile (A-frag reuse x4).
__global__ void __launch_bounds__(256)
k_gemm(const _Float16* __restrict__ A, const _Float16* __restrict__ Bt,
       int M, int N, int K,
       const float* __restrict__ bias, int do_relu,
       const float* __restrict__ resid,
       float* __restrict__ outF, _Float16* __restrict__ outH) {
  int wave = (blockIdx.x * blockDim.x + threadIdx.x) >> 5;
  int ng = N >> 6;
  if (wave >= (M >> 4) * ng) return;      // wave-uniform; EXEC stays all-ones
  int mt = wave / ng, cg = wave - mt * ng;
  int lane = threadIdx.x & 31;
  v8f acc[4];
  v8f z = {};
#pragma unroll
  for (int j = 0; j < 4; ++j) acc[j] = z;
  for (int k0 = 0; k0 < K; k0 += 32) {
    v16h af = load_frag(A, K, mt * 16, k0);
#pragma unroll
    for (int j = 0; j < 4; ++j) {
      v16h bf = load_frag(Bt, K, cg * 64 + j * 16, k0);
      acc[j] = wmma16(af, bf, acc[j]);
    }
  }
  int l15 = lane & 15;
  int rbase = (lane & 16) >> 1;
#pragma unroll
  for (int j = 0; j < 4; ++j) {
#pragma unroll
    for (int r = 0; r < 8; ++r) {
      int row = mt * 16 + r + rbase;
      int col = cg * 64 + j * 16 + l15;
      float v = acc[j][r];
      if (bias)    v += bias[col];
      if (do_relu) v  = fmaxf(v, 0.0f);
      if (resid)   v += resid[(size_t)row * N + col];
      if (outF) outF[(size_t)row * N + col] = v;
      if (outH) outH[(size_t)row * N + col] = (_Float16)v;
    }
  }
}

// ---------------- row LayerNorm over 256 cols (one wave per row) ----------
__global__ void __launch_bounds__(256)
k_ln(const float* __restrict__ in, const float* __restrict__ g,
     const float* __restrict__ b, float* __restrict__ outF,
     _Float16* __restrict__ outH, int rows) {
  int row = (blockIdx.x * blockDim.x + threadIdx.x) >> 5;
  if (row >= rows) return;
  int lane = threadIdx.x & 31;
  const float* x = in + (size_t)row * HD;
  float v[8], s = 0.f, sq = 0.f;
#pragma unroll
  for (int t = 0; t < 8; ++t) {
    v[t] = x[t * 32 + lane];
    s += v[t]; sq += v[t] * v[t];
  }
#pragma unroll
  for (int m = 1; m <= 16; m <<= 1) { s += __shfl_xor(s, m); sq += __shfl_xor(sq, m); }
  float mean = s * (1.0f / HD);
  float var  = sq * (1.0f / HD) - mean * mean;
  float rstd = rsqrtf(var + 1e-5f);
#pragma unroll
  for (int t = 0; t < 8; ++t) {
    int c = t * 32 + lane;
    float y = (v[t] - mean) * rstd * g[c] + b[c];
    outF[(size_t)row * HD + c] = y;
    if (outH) outH[(size_t)row * HD + c] = (_Float16)y;
  }
}

// ---------------- host-side orchestration ----------------
extern "C" void kernel_launch(void* const* d_in, const int* in_sizes, int n_in,
                              void* d_out, int out_size, void* d_ws, size_t ws_size,
                              hipStream_t stream) {
  const float* X   = (const float*)d_in[0];
  const float* Wq  = (const float*)d_in[1];
  const float* Wk  = (const float*)d_in[2];
  const float* Wv  = (const float*)d_in[3];
  const float* Wo  = (const float*)d_in[4];
  const float* w1  = (const float*)d_in[5];
  const float* b1  = (const float*)d_in[6];
  const float* w2  = (const float*)d_in[7];
  const float* b2  = (const float*)d_in[8];
  const float* g1  = (const float*)d_in[9];
  const float* be1 = (const float*)d_in[10];
  const float* g2  = (const float*)d_in[11];
  const float* be2 = (const float*)d_in[12];
  const int*   adj = (const int*)d_in[13];

  char* w = (char*)d_ws;
  const size_t MB = 1u << 20;
  _Float16* Xh   = (_Float16*)(w + 0 * MB);    // 8 MB   [live: qkv]
  _Float16* Qb   = (_Float16*)(w + 8 * MB);    // 8 MB   [qkv -> attn]
  _Float16* Kb   = (_Float16*)(w + 16 * MB);   // 8 MB   [qkv -> attn]
  _Float16* Vtb  = (_Float16*)(w + 24 * MB);   // 8 MB   [qkv -> attn]
  _Float16* outP = (_Float16*)(w + 32 * MB);   // 8 MB   [attn -> Wo gemm]
  _Float16* WqT  = (_Float16*)(w + 40 * MB);         // 128 KB each
  _Float16* WkT  = (_Float16*)(w + 40 * MB + 128 * 1024);
  _Float16* WvT  = (_Float16*)(w + 40 * MB + 256 * 1024);
  _Float16* WoT  = (_Float16*)(w + 40 * MB + 384 * 1024);
  _Float16* w1T  = (_Float16*)(w + 40 * MB + 512 * 1024); // 256 KB
  _Float16* w2T  = (_Float16*)(w + 40 * MB + 768 * 1024); // 256 KB
  float*    Y    = (float*)   (w + 41 * MB);   // 16 MB  [ln1 -> end]
  _Float16* Yh   = (_Float16*)(w + 57 * MB);   // 8 MB   [ln1 -> ffn1]
  // lifetime-overlapped temporaries:
  float*    T1   = (float*)   (w + 8 * MB);    // 16 MB over Qb+Kb  [wo -> ln1]
  _Float16* Fm   = (_Float16*)(w + 8 * MB);    // 16 MB over Qb+Kb  [ffn1 -> ffn2]
  float*    T2   = (float*)   (w + 24 * MB);   // 16 MB over Vtb+outP [ffn2 -> ln2]

  // 1) precision conversion + weight pre-transposition (B operands as Bt[n][k])
  k_f32_to_f16<<<(TOK * HD + 255) / 256, 256, 0, stream>>>(X, Xh, TOK * HD);
  k_transpose_to_f16<<<(HD * HD + 255) / 256, 256, 0, stream>>>(Wq, WqT, HD, HD);
  k_transpose_to_f16<<<(HD * HD + 255) / 256, 256, 0, stream>>>(Wk, WkT, HD, HD);
  k_transpose_to_f16<<<(HD * HD + 255) / 256, 256, 0, stream>>>(Wv, WvT, HD, HD);
  k_transpose_to_f16<<<(HD * HD + 255) / 256, 256, 0, stream>>>(Wo, WoT, HD, HD);
  k_transpose_to_f16<<<(HD * FFND + 255) / 256, 256, 0, stream>>>(w1, w1T, HD, FFND);
  k_transpose_to_f16<<<(FFND * HD + 255) / 256, 256, 0, stream>>>(w2, w2T, FFND, HD);

  // 2) fused QKV projection with head-split permutation (16384 waves)
  k_qkv<<<2048, 256, 0, stream>>>(Xh, WqT, WkT, WvT, Qb, Kb, Vtb);

  // 3) masked attention, permuted f16 output (128 slices x 8 blocks)
  k_attn<<<1024, 256, 0, stream>>>(Qb, Kb, Vtb, adj, outP);

  // 4) attn_out = relu(outP @ Wo) + X   (4096 waves)
  k_gemm<<<512, 256, 0, stream>>>(outP, WoT, TOK, HD, HD,
                                  nullptr, 1, X, T1, nullptr);
  // 5) Y = LN(attn_out + X)
  k_ln<<<2048, 256, 0, stream>>>(T1, g1, be1, Y, Yh, TOK);

  // 6) F = relu(Y @ w1 + b1)            (8192 waves)
  k_gemm<<<1024, 256, 0, stream>>>(Yh, w1T, TOK, FFND, HD,
                                   b1, 1, nullptr, nullptr, Fm);
  // 7) T2 = F @ w2 + b2 + Y             (4096 waves)
  k_gemm<<<512, 256, 0, stream>>>(Fm, w2T, TOK, HD, FFND,
                                  b2, 0, Y, T2, nullptr);
  // 8) out = LN(T2)
  k_ln<<<2048, 256, 0, stream>>>(T2, g2, be2, (float*)d_out, nullptr, TOK);
}
